// ConnectivityLoss_37855841747241
// MI455X (gfx1250) — matrix operand order repaired and checked by reading, not compiled, not run on hardware
//
#include <hip/hip_runtime.h>
#include <hip/hip_bf16.h>

// CDNA5 / gfx1250: wave32, WMMA bf16 16x16x32 (f32 accumulate).
typedef __attribute__((ext_vector_type(16))) __bf16 v16bf;
typedef __attribute__((ext_vector_type(2)))  __bf16 v2bf;
typedef __attribute__((ext_vector_type(8)))  float  v8f;
typedef __attribute__((ext_vector_type(2)))  float  v2f;
typedef __attribute__((ext_vector_type(4)))  float  v4f;

#define NB   64
#define STR  72               // bf16 elements per LDS row: 144B rows (16B aligned, bank-conflict-free)
#define MATB (NB * STR * 2)   // 9216 bytes per bf16 64x64 matrix

// f32 pair -> packed bf16 bits (native v_cvt, RTNE) -- replaces manual bit trick
__device__ __forceinline__ unsigned int pk_bf2(float lo, float hi) {
  v2f pf = {lo, hi};
  v2bf h = __builtin_convertvector(pf, v2bf);
  unsigned int u;
  __builtin_memcpy(&u, &h, 4);
  return u;
}
__device__ __forceinline__ unsigned short bfbits(float f) {
  __bf16 h = (__bf16)f;
  unsigned short u;
  __builtin_memcpy(&u, &h, 2);
  return u;
}
__device__ __forceinline__ float bf2f(unsigned short h) {
  return __uint_as_float(((unsigned int)h) << 16);
}

// 64x64x64 matmul  D = X * Y  in bf16 with f32 accumulation.
// X row-major, Y supplied as its transpose YT (row-major). Writes D row-major
// and optionally DT = D^T (packed b128 stores: 8 consecutive M per lane).
// 8 waves, 2 output 16x16 tiles per wave (same mi -> shared A-fragments),
// K-loop = 2 steps of K=32.
__device__ __forceinline__ void mm64(unsigned short* __restrict__ D,
                                     unsigned short* __restrict__ DT,
                                     const unsigned short* __restrict__ X,
                                     const unsigned short* __restrict__ YT) {
  const int tid  = threadIdx.x;
  const int wave = tid >> 5;
  const int lane = tid & 31;
  const int g    = lane >> 4;   // lane group (0: lanes 0-15, 1: lanes 16-31)
  const int lm   = lane & 15;
#pragma unroll
  for (int j = 0; j < 2; ++j) {
    const int t  = wave * 2 + j;      // tile id 0..15
    const int mi = t >> 2;
    const int ni = t & 3;
    v8f c = {};
#pragma unroll
    for (int k0 = 0; k0 < 64; k0 += 32) {
      v16bf a, b;
      unsigned int* au = reinterpret_cast<unsigned int*>(&a);
      unsigned int* bu = reinterpret_cast<unsigned int*>(&b);
      const int m = mi * 16 + lm;
      const int n = ni * 16 + lm;
#pragma unroll
      for (int v = 0; v < 8; ++v) {
        // ISA 16-bit A 16x32 layout: dword v holds K = ka, ka+1
        const int ka = k0 + 2 * v + ((v >= 4) ? 8 : 0) + 8 * g;
        // ISA B 32x16 layout: dword v holds K = kb, kb+1 for column n
        const int kb = k0 + 2 * v + ((v >= 4) ? 8 : 0) + 8 * g;
        au[v] = *reinterpret_cast<const unsigned int*>(X  + m * STR + ka);
        bu[v] = *reinterpret_cast<const unsigned int*>(YT + n * STR + kb);
      }
      c = __builtin_amdgcn_wmma_f32_16x16x32_bf16(false, a, false, b,
                                                  (short)0, c, false, false);
    }
    // D layout: dword r -> element (M = mi*16 + r + 8g, N = ni*16 + lm)
    unsigned int pk[4];
#pragma unroll
    for (int q = 0; q < 4; ++q) {
      const unsigned int u = pk_bf2(c[2 * q], c[2 * q + 1]);  // one packed cvt
      pk[q] = u;
      D[(mi * 16 + 2 * q     + 8 * g) * STR + ni * 16 + lm] =
          (unsigned short)(u & 0xffffu);                       // ds_store_b16
      D[(mi * 16 + 2 * q + 1 + 8 * g) * STR + ni * 16 + lm] =
          (unsigned short)(u >> 16);                           // ds_store_b16_d16_hi
    }
    if (DT) {
      uint4 pkv;
      pkv.x = pk[0]; pkv.y = pk[1]; pkv.z = pk[2]; pkv.w = pk[3];
      // transposed row ni*16+lm, cols [mi*16 + 8g, +8): 16B-aligned b128 store
      *reinterpret_cast<uint4*>(DT + (ni * 16 + lm) * STR + mi * 16 + 8 * g) = pkv;
    }
  }
}

__global__ void zero_out_kernel(float* o, int n) {
  int i = blockIdx.x * blockDim.x + threadIdx.x;
  if (i < n) o[i] = 0.0f;
}

__global__ void __launch_bounds__(256)
connectivity_loss_kernel(const float* __restrict__ logits,
                         const int* __restrict__ types,
                         float* __restrict__ out, int batches) {
  // LDS pool: 6 bf16 matrices + f32 scratch (overlapping) + deg + labels + acc
  __shared__ __align__(16) unsigned char smem[6 * MATB + 256 + 256 + 16];
  unsigned short* S   = (unsigned short*)(smem + 0 * MATB); // s  (row-major)
  unsigned short* STm = (unsigned short*)(smem + 1 * MATB); // s^T
  unsigned short* R   = (unsigned short*)(smem + 2 * MATB); // r  (row-major)
  unsigned short* Rb  = (unsigned short*)(smem + 3 * MATB); // r ping-pong
  unsigned short* T   = (unsigned short*)(smem + 4 * MATB); // temp s^2
  unsigned short* TT  = (unsigned short*)(smem + 5 * MATB); // temp (s^2)^T
  float* Pf  = (float*)(smem + 4 * MATB);        // f32 p scratch; overlaps T/TT
                                                 // (dead before first matmul)
  float* deg = (float*)(smem + 6 * MATB);
  int*   lab = (int*)  (smem + 6 * MATB + 256);
  float* acc = (float*)(smem + 6 * MATB + 512);

  const int b   = blockIdx.x;
  const int tid = threadIdx.x;

  if (tid < NB) lab[tid] = types[b * NB + tid];
  if (tid == 0) acc[0] = 0.0f;

  // Each thread: one quarter-row of the 64x64 tile.
  // Streaming single-use data -> non-temporal b128 loads.
  const int row = tid >> 2;
  const int c0  = (tid & 3) * 16;
  const float* src = logits + (size_t)b * NB * NB + row * NB + c0;
#pragma unroll
  for (int q = 0; q < 4; ++q) {
    v4f x = __builtin_nontemporal_load(
        reinterpret_cast<const v4f*>(src + 4 * q));
#pragma unroll
    for (int e = 0; e < 4; ++e) {
      const int col  = c0 + 4 * q + e;
      const float p  = 1.0f / (1.0f + __expf(-x[e]));
      Pf[row * 65 + col] = p;                       // exact f32 for degrees
      const unsigned short h = bfbits(p + ((row == col) ? 1.0f : 0.0f)); // A = p+I
      S  [row * STR + col] = h;
      STm[col * STR + row] = h;
      R  [row * STR + col] = h;                     // r starts as A
    }
  }
  __syncthreads();

  // deg[i] = row-sum + col-sum of p, in f32 (stride 65 -> conflict-free cols)
  if (tid < NB) {
    float s = 0.0f;
    for (int jj = 0; jj < NB; ++jj) s += Pf[tid * 65 + jj] + Pf[jj * 65 + tid];
    deg[tid] = s;
  }
  __syncthreads();

  // A^63 by squaring: r=s=A; 5x { s = s*s; r = r*s }  -> r = A^(1+2+4+8+16+32)
  unsigned short *ps = S, *pst = STm, *pr = R, *prb = Rb, *pt = T, *ptt = TT;
#pragma unroll 1
  for (int it = 0; it < 5; ++it) {
    mm64(pt, ptt, ps, pst);        // t = s*s (plus transpose for next B-operand)
    __syncthreads();
    mm64(prb, nullptr, pr, ptt);   // r' = r * s^2
    __syncthreads();
    unsigned short* tmp;
    tmp = ps;  ps  = pt;  pt  = tmp;
    tmp = pst; pst = ptt; ptt = tmp;
    tmp = pr;  pr  = prb; prb = tmp;
  }
  // pr = A^63 (bf16, row-major)

  // graph loss: sum relu(0.1 - A_pow * valid_i*valid_j) over all pairs
  float gsum = 0.0f;
  const int vi = (lab[row] < 4);
#pragma unroll
  for (int e = 0; e < 16; ++e) {
    const int col  = c0 + e;
    const float ap = bf2f(pr[row * STR + col]);
    const float vp = (vi && (lab[col] < 4)) ? ap : 0.0f;
    gsum += fmaxf(0.1f - vp, 0.0f);
  }
  atomicAdd(&acc[0], gsum);        // ds_add_f32
  __syncthreads();

  if (tid == 0) {
    float pen1 = 0.0f, pen2 = 0.0f, iso = 0.0f;
    int f1 = 0, f2 = 0;
    for (int i = 0; i < NB; ++i) {
      const int L = lab[i];
      if (!f1 && L == 1) { f1 = 1; pen1 = fmaxf(1.0f - deg[i], 0.0f); }
      if (!f2 && L == 2) { f2 = 1; pen2 = fmaxf(1.0f - deg[i], 0.0f); }
      if (L < 4) iso += fmaxf(0.5f - deg[i], 0.0f);
    }
    const float Bf = (float)batches;
    const float contrib = 10.0f * pen1 / Bf
                        +  5.0f * pen2 / Bf
                        +  3.0f * acc[0] / (Bf * 64.0f * 64.0f)
                        +  2.0f * iso / (Bf * 64.0f);
    atomicAdd(out, contrib);       // global_atomic_add_f32
  }
}

extern "C" void kernel_launch(void* const* d_in, const int* in_sizes, int n_in,
                              void* d_out, int out_size, void* d_ws, size_t ws_size,
                              hipStream_t stream) {
  const float* logits = (const float*)d_in[0];   // [B, 64, 64] f32
  const int*   types  = (const int*)d_in[1];     // [B, 64] int
  float* out = (float*)d_out;                    // scalar
  const int batches = in_sizes[1] / NB;

  zero_out_kernel<<<1, 64, 0, stream>>>(out, out_size);
  connectivity_loss_kernel<<<batches, 256, 0, stream>>>(logits, types, out, batches);
}
